// myModel_3599182594341
// MI455X (gfx1250) — matrix operand order repaired and checked by reading, not compiled, not run on hardware
//
#include <hip/hip_runtime.h>
#include <stdint.h>

#define H 1024
#define V 128
#define BB 256
#define SS 512

typedef __attribute__((ext_vector_type(16))) __bf16 v16bf;
typedef __attribute__((ext_vector_type(8)))  float  v8f;

union Frag16 { v16bf v; uint4 q[2]; };

__device__ __forceinline__ unsigned short f32_to_bf16(float f) {
    union { float f; unsigned u; } c; c.f = f;
    unsigned r = 0x7FFFu + ((c.u >> 16) & 1u);      // round-to-nearest-even
    return (unsigned short)((c.u + r) >> 16);
}
__device__ __forceinline__ float bf16_to_f32(unsigned short h) {
    union { unsigned u; float f; } c; c.u = ((unsigned)h) << 16;
    return c.f;
}

// CDNA5 hardware tanh (TRANS op). v_nop covers the documented TRANS-op
// result hazard since the compiler cannot schedule around asm outputs.
__device__ __forceinline__ float hw_tanhf(float x) {
    float r;
    asm volatile("v_tanh_f32 %0, %1\n\tv_nop" : "=v"(r) : "v"(x));
    return r;
}

// emb[t][h] = W_ih[h][t] + b_ih[h] + b_hh[h]   (fp32, 128x1024, L2-resident)
__global__ void pack_emb_kernel(const float* __restrict__ W_ih,
                                const float* __restrict__ b_ih,
                                const float* __restrict__ b_hh,
                                float* __restrict__ emb) {
    int i = blockIdx.x * 256 + threadIdx.x;      // i in [0, V*H)
    int t = i >> 10, h = i & (H - 1);
    emb[i] = W_ih[h * V + t] + b_ih[h] + b_hh[h];
}

// Pack a row-major [N][K] fp32 weight (used as W.T in h@W.T) into bf16 WMMA
// B-fragments (K=32, N=16 per fragment), *kc-major* so one kc iteration of a
// workgroup touches one contiguous span of L2:
//   frag f = kc*ntiles + nt ; dst[f*512 + lane*16 + j] = bf16( W[n][k] )
//   n = nt*16 + (lane&15) ; k = kc*32 + (j<8 ? 8*(lane>>4)+j : 16+8*(lane>>4)+(j-8))
__global__ void pack_bfrag_kernel(const float* __restrict__ W,
                                  unsigned short* __restrict__ dst, int ntiles) {
    int i    = blockIdx.x * 256 + threadIdx.x;
    int j    = i & 15;
    int lane = (i >> 4) & 31;
    int f    = i >> 9;               // fragment index = kc*ntiles + nt
    int nt   = f % ntiles;
    int kc   = f / ntiles;
    int n = nt * 16 + (lane & 15);
    int k = kc * 32 + ((j < 8) ? 8 * (lane >> 4) + j : 16 + 8 * (lane >> 4) + (j - 8));
    dst[i] = f32_to_bf16(W[n * H + k]);
}

// Persistent RNN kernel: one workgroup per 16-row batch tile, 8 waves.
// Waves split the 1024 hidden columns (recurrence) and the 128 vocab
// columns (fused output projection). h double-buffered in LDS as bf16.
// A laundered zero offset (asm-defined SGPR) is added to the weight bases
// each step: loads stay GLOBAL (kernarg provenance kept -> no FLAT/DScnt
// coupling), but the compiler cannot hoist s-invariant weight fragments
// into registers (which caused scratch spills). Registers go to a deep
// double-buffered pipeline: 16 global_load_b128 in flight over 8 WMMAs.
__global__ __launch_bounds__(256, 1)
void rnn_kernel(const int* __restrict__ x, const float* __restrict__ state,
                const float* __restrict__ emb,
                const unsigned short* __restrict__ whh,
                const unsigned short* __restrict__ wout,
                const float* __restrict__ bout,
                float* __restrict__ y, float* __restrict__ hlast) {
    __shared__ unsigned short hbuf[2][16 * H];    // 64 KB of 320 KB LDS

    const int tid   = threadIdx.x;
    const int wave  = tid >> 5;
    const int lane  = tid & 31;
    const int b0    = blockIdx.x * 16;
    const int lhalf = lane >> 4;                  // 0: rows M=0..7 / K-lo ; 1: M=8..15 / K-hi
    const int n16   = lane & 15;

    // init h from `state` (fp32 -> bf16)
    for (int i = tid; i < 16 * H; i += 256)
        hbuf[0][i] = f32_to_bf16(state[(b0 + (i >> 10)) * H + (i & (H - 1))]);
    __syncthreads();

    const int   ntb   = wave * 8;                 // this wave's 8 hidden N-tiles
    const int   aoff0 = lhalf ? 8 : 0;            // A-fragment K sub-offset (ISA 16-bit A layout)
    const float pbias = bout[wave * 16 + n16];    // this wave's vocab tile bias

    for (int s = 0; s < SS; ++s) {
        const int cur = s & 1, nxt = cur ^ 1;

        // Laundered zero: defeats cross-step hoisting of weight fragments
        // while keeping global (not flat) addressing on the weight loads.
        size_t lz = 0;
        asm volatile("" : "+s"(lz));

        // tokens of the 8 batch rows this lane-half contributes to (L2 hits)
        int tok[8];
#pragma unroll
        for (int r = 0; r < 8; ++r)
            tok[r] = x[(b0 + r + lhalf * 8) * SS + s];

        // C := pre  (embedding gather + biases, directly in accumulator layout)
        v8f acc[8];
#pragma unroll
        for (int t = 0; t < 8; ++t) {
            const int ng = (ntb + t) * 16 + n16;
#pragma unroll
            for (int r = 0; r < 8; ++r)
                acc[t][r] = emb[tok[r] * H + ng];
        }

        // this wave's fragment base: frag(t, kc) at wb + (kc*64 + t)*512
        const unsigned short* wb = whh + lz + (size_t)ntb * 512 + (size_t)lane * 16;

        // acc += h @ W_hh.T : 32 K-chunks, 8 independent WMMA chains.
        // B fragments double-buffered in registers, prefetched 1 kc ahead.
        v16bf bA[8], bB[8];
#pragma unroll
        for (int t = 0; t < 8; ++t)
            bA[t] = *(const v16bf*)(wb + t * 512);

        for (int kc = 0; kc < 32; kc += 2) {
            {
                const unsigned short* w1 = wb + (size_t)(kc + 1) * 64 * 512;
#pragma unroll
                for (int t = 0; t < 8; ++t)
                    bB[t] = *(const v16bf*)(w1 + t * 512);
            }
            {
                Frag16 a;
                const unsigned short* ap = &hbuf[cur][n16 * H + kc * 32 + aoff0];
                a.q[0] = *(const uint4*)(ap);
                a.q[1] = *(const uint4*)(ap + 16);
#pragma unroll
                for (int t = 0; t < 8; ++t)
                    acc[t] = __builtin_amdgcn_wmma_f32_16x16x32_bf16(
                                 false, a.v, false, bA[t], (short)0, acc[t], false, false);
            }
            if (kc + 2 < 32) {
                const unsigned short* w2 = wb + (size_t)(kc + 2) * 64 * 512;
#pragma unroll
                for (int t = 0; t < 8; ++t)
                    bA[t] = *(const v16bf*)(w2 + t * 512);
            }
            {
                Frag16 a;
                const unsigned short* ap = &hbuf[cur][n16 * H + (kc + 1) * 32 + aoff0];
                a.q[0] = *(const uint4*)(ap);
                a.q[1] = *(const uint4*)(ap + 16);
#pragma unroll
                for (int t = 0; t < 8; ++t)
                    acc[t] = __builtin_amdgcn_wmma_f32_16x16x32_bf16(
                                 false, a.v, false, bB[t], (short)0, acc[t], false, false);
            }
        }

        // h_new = tanh(acc) -> bf16 into the other LDS buffer (HW v_tanh_f32)
#pragma unroll
        for (int t = 0; t < 8; ++t) {
            const int ng = (ntb + t) * 16 + n16;
#pragma unroll
            for (int r = 0; r < 8; ++r) {
                const int m = r + lhalf * 8;
                hbuf[nxt][m * H + ng] = f32_to_bf16(hw_tanhf(acc[t][r]));
            }
        }
        __syncthreads();   // one barrier per step (double-buffered h)

        // fused projection: y_s(tile) = h_new @ W_out.T + b_out ; wave = V-tile.
        // 4 partial accumulators; B double-buffered in groups of 4 kc.
        const unsigned short* ob = wout + lz + (size_t)wave * 512 + (size_t)lane * 16;
        // frag(kc) at ob + kc*8*512

        v8f oc[4];
#pragma unroll
        for (int r = 0; r < 8; ++r) {
            oc[0][r] = pbias;
            oc[1][r] = 0.0f; oc[2][r] = 0.0f; oc[3][r] = 0.0f;
        }
        v16bf pA[4], pB[4];
#pragma unroll
        for (int p = 0; p < 4; ++p)
            pA[p] = *(const v16bf*)(ob + (size_t)p * 8 * 512);

        for (int kq = 0; kq < 8; kq += 2) {
#pragma unroll
            for (int p = 0; p < 4; ++p)
                pB[p] = *(const v16bf*)(ob + (size_t)((kq + 1) * 4 + p) * 8 * 512);
#pragma unroll
            for (int p = 0; p < 4; ++p) {
                const int kc = kq * 4 + p;
                Frag16 a;
                const unsigned short* ap = &hbuf[nxt][n16 * H + kc * 32 + aoff0];
                a.q[0] = *(const uint4*)(ap);
                a.q[1] = *(const uint4*)(ap + 16);
                oc[p] = __builtin_amdgcn_wmma_f32_16x16x32_bf16(
                            false, a.v, false, pA[p], (short)0, oc[p], false, false);
            }
            if (kq + 2 < 8) {
#pragma unroll
                for (int p = 0; p < 4; ++p)
                    pA[p] = *(const v16bf*)(ob + (size_t)((kq + 2) * 4 + p) * 8 * 512);
            }
#pragma unroll
            for (int p = 0; p < 4; ++p) {
                const int kc = (kq + 1) * 4 + p;
                Frag16 a;
                const unsigned short* ap = &hbuf[nxt][n16 * H + kc * 32 + aoff0];
                a.q[0] = *(const uint4*)(ap);
                a.q[1] = *(const uint4*)(ap + 16);
                oc[p] = __builtin_amdgcn_wmma_f32_16x16x32_bf16(
                            false, a.v, false, pB[p], (short)0, oc[p], false, false);
            }
        }
#pragma unroll
        for (int r = 0; r < 8; ++r) {
            const int m = r + lhalf * 8;
            y[((size_t)s * BB + (b0 + m)) * V + wave * 16 + n16] =
                (oc[0][r] + oc[1][r]) + (oc[2][r] + oc[3][r]);
        }
    }
    __syncthreads();

    // h_last (final state sits in hbuf[0] since S is even)
    for (int i = tid; i < 16 * H; i += 256)
        hlast[(b0 + (i >> 10)) * H + (i & (H - 1))] = bf16_to_f32(hbuf[0][i]);
}

extern "C" void kernel_launch(void* const* d_in, const int* in_sizes, int n_in,
                              void* d_out, int out_size, void* d_ws, size_t ws_size,
                              hipStream_t stream) {
    const int*   x     = (const int*)d_in[0];
    const float* state = (const float*)d_in[1];
    const float* W_ih  = (const float*)d_in[2];
    const float* W_hh  = (const float*)d_in[3];
    const float* b_ih  = (const float*)d_in[4];
    const float* b_hh  = (const float*)d_in[5];
    const float* W_out = (const float*)d_in[6];
    const float* b_out = (const float*)d_in[7];

    float* y     = (float*)d_out;
    float* hlast = y + (size_t)SS * BB * V;

    char* ws = (char*)d_ws;
    float*          emb  = (float*)ws;                           // 512 KB
    unsigned short* whh  = (unsigned short*)(ws + 524288);       // 2 MB  (64 N-tiles, kc-major)
    unsigned short* wout = (unsigned short*)(ws + 2621440);      // 256 KB (8 N-tiles, kc-major)

    pack_emb_kernel  <<<512,  256, 0, stream>>>(W_ih, b_ih, b_hh, emb);
    pack_bfrag_kernel<<<4096, 256, 0, stream>>>(W_hh, whh, 64);
    pack_bfrag_kernel<<<512,  256, 0, stream>>>(W_out, wout, 8);
    rnn_kernel       <<<16,   256, 0, stream>>>(x, state, emb, whh, wout, b_out, y, hlast);
}
